// Graph_CNN_Feat_Mesh_10015863734925
// MI455X (gfx1250) — matrix operand order repaired and checked by reading, not compiled, not run on hardware
//
#include <hip/hip_runtime.h>
#include <cstddef>

// ---------------------------------------------------------------------------
// Problem constants (from the reference)
// ---------------------------------------------------------------------------
#define BATCH 256
#define CIN   2048
#define V0    80
#define V1    320
#define V2    1280
#define DEG   8
#define BN_EPS 1e-5f

typedef __attribute__((ext_vector_type(16))) __bf16 bf16x16;
typedef __attribute__((ext_vector_type(8)))  __bf16 bf16x8;
typedef __attribute__((ext_vector_type(8)))  float  f32x8;

// ---------------------------------------------------------------------------
// Elementwise: fp32 -> bf16 convert
// ---------------------------------------------------------------------------
__global__ void k_f32_to_bf16(const float* __restrict__ src,
                              __bf16* __restrict__ dst, size_t n) {
    size_t i = (size_t)blockIdx.x * blockDim.x + threadIdx.x;
    if (i < n) dst[i] = (__bf16)src[i];
}

// Convert weight [N,K] fp32 -> [Npad,K] bf16, zero-padding rows N..Npad-1
__global__ void k_convert_pad_w(const float* __restrict__ src,
                                __bf16* __restrict__ dst,
                                int N, int Npad, int K) {
    size_t i = (size_t)blockIdx.x * blockDim.x + threadIdx.x;
    size_t total = (size_t)Npad * K;
    if (i >= total) return;
    int row = (int)(i / K);
    float v = (row < N) ? src[i] : 0.0f;
    dst[i] = (__bf16)v;
}

// ---------------------------------------------------------------------------
// Generic GEMM: C[M,N] = A[M,K](bf16) @ W[Npad,K](bf16)^T + bias, optional relu
// M % 128 == 0, Npad % 16 == 0, K % 32 == 0.
// out_mode 0: C[r*N + n] ; out_mode 1: C[((r%Bdim)*Vdim + r/Bdim)*N + n]
// One wave per 16x16 output tile, 8 waves (256 threads) per block.
// ---------------------------------------------------------------------------
__global__ __launch_bounds__(256)
void k_gemm_bf16_wmma(const __bf16* __restrict__ A,
                      const __bf16* __restrict__ W,
                      const float* __restrict__ bias,
                      float* __restrict__ C,
                      int M, int N, int K,
                      int do_relu, int out_mode, int Vdim, int Bdim) {
    const int lane   = threadIdx.x & 31;
    const int wave   = threadIdx.x >> 5;
    const int lane15 = lane & 15;
    const int hi     = (lane >= 16) ? 1 : 0;

    const int m0   = (blockIdx.x * 8 + wave) * 16;   // row tile base
    const int ncol = blockIdx.y * 16 + lane15;       // this lane's output column

    const __bf16* __restrict__ Arow = A + (size_t)(m0 + lane15) * K;
    const __bf16* __restrict__ Wrow = W + (size_t)ncol * K;

    f32x8 acc = {};

    for (int kk = 0; kk < K; kk += 32) {
        // A fragment: contiguous k order (k = 16*hi + elem), same as B below.
        //   lanes 0-15 : elems 0..15 -> K = kk+0..15
        //   lanes 16-31: elems 0..15 -> K = kk+16..31
        const int kbase = kk + (hi ? 16 : 0);
        bf16x8 alo = *(const bf16x8*)(Arow + kbase);
        bf16x8 ahi = *(const bf16x8*)(Arow + kbase + 8);
        bf16x16 a;
#pragma unroll
        for (int i = 0; i < 8; ++i) { a[i] = alo[i]; a[i + 8] = ahi[i]; }

        // B fragment: 32x16, lane = column; lanes 0-15 hold K=kk+0..15,
        // lanes 16-31 hold K=kk+16..31 (contiguous along K of W row).
        bf16x16 b = *(const bf16x16*)(Wrow + kk + (hi ? 16 : 0));

        __builtin_prefetch(Arow + kk + 128, 0, 0);

        acc = __builtin_amdgcn_wmma_f32_16x16x32_bf16(
            /*neg_a=*/false, a, /*neg_b=*/false, b,
            /*c_mod=*/(short)0, acc, /*reuse_a=*/false, /*reuse_b=*/false);
    }

    // Store. C/D layout: lane L holds column L&15; VGPR j holds row j (+8 if hi).
    if (ncol < N) {
        const float bv = bias[ncol];
#pragma unroll
        for (int j = 0; j < 8; ++j) {
            int r = m0 + j + (hi ? 8 : 0);
            float v = acc[j] + bv;
            if (do_relu) v = fmaxf(v, 0.0f);
            size_t idx;
            if (out_mode == 0) {
                idx = (size_t)r * N + ncol;
            } else {
                int bb = r % Bdim;
                int vv = r / Bdim;
                idx = ((size_t)bb * Vdim + vv) * N + ncol;
            }
            C[idx] = v;
        }
    }
}

// ---------------------------------------------------------------------------
// SpMM over structured Laplacian: y[v,:] = diag(v)*x[v,:] + sum_j vals[v*8+j]*x[cols[v*8+j],:]
// N4 = N/4 float4 columns. Second variant: y = 2*(L@x1) - xt.
// ---------------------------------------------------------------------------
__global__ void k_spmm(const int* __restrict__ cols, const float* __restrict__ vals,
                       const float* __restrict__ x, float* __restrict__ y,
                       int V, int N4) {
    size_t gid = (size_t)blockIdx.x * blockDim.x + threadIdx.x;
    size_t total = (size_t)V * N4;
    if (gid >= total) return;
    int v  = (int)(gid / N4);
    int c4 = (int)(gid - (size_t)v * N4);
    const int N = N4 * 4;
    float dv = vals[(size_t)DEG * V + v];
    const float4* xv = (const float4*)(x + (size_t)v * N) + c4;
    float4 a = *xv;
    float4 s;
    s.x = dv * a.x; s.y = dv * a.y; s.z = dv * a.z; s.w = dv * a.w;
#pragma unroll
    for (int j = 0; j < DEG; ++j) {
        int   col = cols[v * DEG + j];
        float w   = vals[v * DEG + j];
        float4 b  = *((const float4*)(x + (size_t)col * N) + c4);
        s.x += w * b.x; s.y += w * b.y; s.z += w * b.z; s.w += w * b.w;
    }
    *((float4*)(y + (size_t)v * N) + c4) = s;
}

__global__ void k_spmm2(const int* __restrict__ cols, const float* __restrict__ vals,
                        const float* __restrict__ x1, const float* __restrict__ xt,
                        float* __restrict__ y, int V, int N4) {
    size_t gid = (size_t)blockIdx.x * blockDim.x + threadIdx.x;
    size_t total = (size_t)V * N4;
    if (gid >= total) return;
    int v  = (int)(gid / N4);
    int c4 = (int)(gid - (size_t)v * N4);
    const int N = N4 * 4;
    float dv = vals[(size_t)DEG * V + v];
    float4 a = *((const float4*)(x1 + (size_t)v * N) + c4);
    float4 s;
    s.x = dv * a.x; s.y = dv * a.y; s.z = dv * a.z; s.w = dv * a.w;
#pragma unroll
    for (int j = 0; j < DEG; ++j) {
        int   col = cols[v * DEG + j];
        float w   = vals[v * DEG + j];
        float4 b  = *((const float4*)(x1 + (size_t)col * N) + c4);
        s.x += w * b.x; s.y += w * b.y; s.z += w * b.z; s.w += w * b.w;
    }
    float4 t = *((const float4*)(xt + (size_t)v * N) + c4);
    s.x = 2.0f * s.x - t.x; s.y = 2.0f * s.y - t.y;
    s.z = 2.0f * s.z - t.z; s.w = 2.0f * s.w - t.w;
    *((float4*)(y + (size_t)v * N) + c4) = s;
}

// ---------------------------------------------------------------------------
// Build xt0 [V1, B*64] from fc2 output h2 [B, V0*64] with x4 vertex upsample.
// ---------------------------------------------------------------------------
__global__ void k_build_xt0(const float* __restrict__ h2, float* __restrict__ xt) {
    size_t i = (size_t)blockIdx.x * blockDim.x + threadIdx.x;
    const size_t total = (size_t)V1 * BATCH * 64;
    if (i >= total) return;
    int v   = (int)(i / (BATCH * 64));
    int rem = (int)(i - (size_t)v * (BATCH * 64));
    int b   = rem / 64;
    int f   = rem - b * 64;
    xt[i] = h2[(size_t)b * (V0 * 64) + (v >> 2) * 64 + f];
}

// Vertex x4 upsample in xt layout: dst[v,:] = src[v>>2,:]
__global__ void k_upsample4(const float* __restrict__ src, float* __restrict__ dst,
                            int V, int rowlen) {
    size_t i = (size_t)blockIdx.x * blockDim.x + threadIdx.x;
    size_t total = (size_t)V * rowlen;
    if (i >= total) return;
    int v = (int)(i / rowlen);
    int c = (int)(i - (size_t)v * rowlen);
    dst[i] = src[(size_t)(v >> 2) * rowlen + c];
}

// ---------------------------------------------------------------------------
// Build xs (bf16) [rows, 3F] with (f,k) ordering, k minor: xs[r, f*3+k] = arr_k[r*F+f]
// ---------------------------------------------------------------------------
__global__ void k_build_xs(const float* __restrict__ x0, const float* __restrict__ x1,
                           const float* __restrict__ x2, __bf16* __restrict__ xs,
                           size_t rows, int F) {
    size_t i = (size_t)blockIdx.x * blockDim.x + threadIdx.x;
    const int K3 = 3 * F;
    size_t total = rows * (size_t)K3;
    if (i >= total) return;
    size_t r = i / K3;
    int c = (int)(i - r * K3);
    int f = c / 3;
    int k = c - f * 3;
    const float* src = (k == 0) ? x0 : (k == 1) ? x1 : x2;
    xs[i] = (__bf16)src[r * F + f];
}

// ---------------------------------------------------------------------------
// BatchNorm (training): per-channel sum/sumsq, then apply (biased var) + relu
// ---------------------------------------------------------------------------
__global__ void k_bn_stats(const float* __restrict__ y, float* __restrict__ stats,
                           size_t rows, int C) {
    __shared__ float sh_s[256];
    __shared__ float sh_q[256];
    int c = blockIdx.x;
    float s = 0.0f, q = 0.0f;
    for (size_t r = threadIdx.x; r < rows; r += blockDim.x) {
        float v = y[r * C + c];
        s += v; q += v * v;
    }
    sh_s[threadIdx.x] = s; sh_q[threadIdx.x] = q;
    __syncthreads();
    for (int off = 128; off > 0; off >>= 1) {
        if ((int)threadIdx.x < off) {
            sh_s[threadIdx.x] += sh_s[threadIdx.x + off];
            sh_q[threadIdx.x] += sh_q[threadIdx.x + off];
        }
        __syncthreads();
    }
    if (threadIdx.x == 0) { stats[2 * c] = sh_s[0]; stats[2 * c + 1] = sh_q[0]; }
}

__global__ void k_bn_apply_relu(const float* __restrict__ y,
                                const float* __restrict__ stats,
                                const float* __restrict__ g,
                                const float* __restrict__ b,
                                float* __restrict__ out,
                                size_t total, int C, float invcnt) {
    size_t i = (size_t)blockIdx.x * blockDim.x + threadIdx.x;
    if (i >= total) return;
    int c = (int)(i % C);
    float mu  = stats[2 * c] * invcnt;
    float var = stats[2 * c + 1] * invcnt - mu * mu;
    float v = (y[i] - mu) * rsqrtf(var + BN_EPS) * g[c] + b[c];
    out[i] = fmaxf(v, 0.0f);
}

// ---------------------------------------------------------------------------
// Host launcher
// ---------------------------------------------------------------------------
static inline int cdiv(size_t a, int b) { return (int)((a + b - 1) / b); }

extern "C" void kernel_launch(void* const* d_in, const int* in_sizes, int n_in,
                              void* d_out, int out_size, void* d_ws, size_t ws_size,
                              hipStream_t stream) {
    (void)in_sizes; (void)n_in; (void)out_size; (void)ws_size;
    const float* x      = (const float*)d_in[0];
    const float* fc1_w  = (const float*)d_in[1];
    const float* fc1_b  = (const float*)d_in[2];
    const float* fc2_w  = (const float*)d_in[3];
    const float* fc2_b  = (const float*)d_in[4];
    const float* cl0_w  = (const float*)d_in[5];
    const float* cl0_b  = (const float*)d_in[6];
    const float* cl1_w  = (const float*)d_in[7];
    const float* cl1_b  = (const float*)d_in[8];
    const float* cl2_w  = (const float*)d_in[9];
    const float* cl2_b  = (const float*)d_in[10];
    const float* cl3_w  = (const float*)d_in[11];
    const float* cl3_b  = (const float*)d_in[12];
    const float* bn0_g  = (const float*)d_in[13];
    const float* bn0_b  = (const float*)d_in[14];
    const float* bn1_g  = (const float*)d_in[15];
    const float* bn1_b  = (const float*)d_in[16];
    const float* bn2_g  = (const float*)d_in[17];
    const float* bn2_b  = (const float*)d_in[18];
    const int*   L1c    = (const int*)d_in[20];
    const float* L1v    = (const float*)d_in[21];
    const int*   L2c    = (const int*)d_in[23];
    const float* L2v    = (const float*)d_in[24];
    float* out = (float*)d_out;

    // Workspace partition (bytes)
    char* ws = (char*)d_ws;
    __bf16* Abf = (__bf16*)(ws);                         // 64 MB bf16 A staging
    __bf16* Wbf = (__bf16*)(ws + 67108864);              //  6 MB bf16 W staging
    float*  F0  = (float*)(ws + 67108864 + 6291456);     // 4x 40 MB fp32 buffers
    float*  F1  = F0 + 10485760;
    float*  F2  = F1 + 10485760;
    float*  F3  = F2 + 10485760;
    float*  ST  = F3 + 10485760;                         // BN stats (sum,sumsq)

    const int T = 256;
    dim3 blk(T);

    // ---------------- FC1: [256,2048] x [512,2048]^T, relu -> F3 ----------
    k_f32_to_bf16<<<cdiv((size_t)BATCH * CIN, T), blk, 0, stream>>>(x, Abf, (size_t)BATCH * CIN);
    k_convert_pad_w<<<cdiv((size_t)512 * CIN, T), blk, 0, stream>>>(fc1_w, Wbf, 512, 512, CIN);
    k_gemm_bf16_wmma<<<dim3(BATCH / 128, 512 / 16), blk, 0, stream>>>(
        Abf, Wbf, fc1_b, F3, BATCH, 512, CIN, 1, 0, 0, 1);

    // ---------------- FC2: [256,512] x [5120,512]^T -> F3 (h2) ------------
    k_f32_to_bf16<<<cdiv((size_t)BATCH * 512, T), blk, 0, stream>>>(F3, Abf, (size_t)BATCH * 512);
    k_convert_pad_w<<<cdiv((size_t)5120 * 512, T), blk, 0, stream>>>(fc2_w, Wbf, 5120, 5120, 512);
    k_gemm_bf16_wmma<<<dim3(BATCH / 128, 5120 / 16), blk, 0, stream>>>(
        Abf, Wbf, fc2_b, F3, BATCH, 5120, 512, 0, 0, 0, 1);

    // ---------------- xt0 [320, B*64] with x4 upsample --------------------
    {
        size_t n = (size_t)V1 * BATCH * 64;
        k_build_xt0<<<cdiv(n, T), blk, 0, stream>>>(F3, F0);
    }

    // ================ Cheby layer 0 @ V=320, F=64 -> 64 ===================
    {
        const int N = BATCH * 64, N4 = N / 4;
        size_t rows = (size_t)V1 * BATCH;
        k_spmm <<<cdiv((size_t)V1 * N4, T), blk, 0, stream>>>(L1c, L1v, F0, F1, V1, N4);
        k_spmm2<<<cdiv((size_t)V1 * N4, T), blk, 0, stream>>>(L1c, L1v, F1, F0, F2, V1, N4);
        k_build_xs<<<cdiv(rows * 192, T), blk, 0, stream>>>(F0, F1, F2, Abf, rows, 64);
        k_convert_pad_w<<<cdiv((size_t)64 * 192, T), blk, 0, stream>>>(cl0_w, Wbf, 64, 64, 192);
        k_gemm_bf16_wmma<<<dim3((int)(rows / 128), 64 / 16), blk, 0, stream>>>(
            Abf, Wbf, cl0_b, F3, (int)rows, 64, 192, 0, 0, 0, 1);
        k_bn_stats<<<64, blk, 0, stream>>>(F3, ST, rows, 64);
        k_bn_apply_relu<<<cdiv(rows * 64, T), blk, 0, stream>>>(
            F3, ST, bn0_g, bn0_b, F0, rows * 64, 64, 1.0f / (float)rows);
    }

    // ================ Cheby layer 1 @ V=320, F=64 -> 32 ===================
    {
        const int N = BATCH * 64, N4 = N / 4;
        size_t rows = (size_t)V1 * BATCH;
        k_spmm <<<cdiv((size_t)V1 * N4, T), blk, 0, stream>>>(L1c, L1v, F0, F1, V1, N4);
        k_spmm2<<<cdiv((size_t)V1 * N4, T), blk, 0, stream>>>(L1c, L1v, F1, F0, F2, V1, N4);
        k_build_xs<<<cdiv(rows * 192, T), blk, 0, stream>>>(F0, F1, F2, Abf, rows, 64);
        k_convert_pad_w<<<cdiv((size_t)32 * 192, T), blk, 0, stream>>>(cl1_w, Wbf, 32, 32, 192);
        k_gemm_bf16_wmma<<<dim3((int)(rows / 128), 32 / 16), blk, 0, stream>>>(
            Abf, Wbf, cl1_b, F3, (int)rows, 32, 192, 0, 0, 0, 1);
        k_bn_stats<<<32, blk, 0, stream>>>(F3, ST, rows, 32);
        k_bn_apply_relu<<<cdiv(rows * 32, T), blk, 0, stream>>>(
            F3, ST, bn1_g, bn1_b, F3, rows * 32, 32, 1.0f / (float)rows);
    }

    // ---------------- x4 upsample to V=1280 in xt layout -> F0 ------------
    k_upsample4<<<cdiv((size_t)V2 * (BATCH * 32), T), blk, 0, stream>>>(F3, F0, V2, BATCH * 32);

    // ================ Cheby layer 2 @ V=1280, F=32 -> 32 ==================
    {
        const int N = BATCH * 32, N4 = N / 4;
        size_t rows = (size_t)V2 * BATCH;
        k_spmm <<<cdiv((size_t)V2 * N4, T), blk, 0, stream>>>(L2c, L2v, F0, F1, V2, N4);
        k_spmm2<<<cdiv((size_t)V2 * N4, T), blk, 0, stream>>>(L2c, L2v, F1, F0, F2, V2, N4);
        k_build_xs<<<cdiv(rows * 96, T), blk, 0, stream>>>(F0, F1, F2, Abf, rows, 32);
        k_convert_pad_w<<<cdiv((size_t)32 * 96, T), blk, 0, stream>>>(cl2_w, Wbf, 32, 32, 96);
        k_gemm_bf16_wmma<<<dim3((int)(rows / 128), 32 / 16), blk, 0, stream>>>(
            Abf, Wbf, cl2_b, F3, (int)rows, 32, 96, 0, 0, 0, 1);
        k_bn_stats<<<32, blk, 0, stream>>>(F3, ST, rows, 32);
        k_bn_apply_relu<<<cdiv(rows * 32, T), blk, 0, stream>>>(
            F3, ST, bn2_g, bn2_b, F3, rows * 32, 32, 1.0f / (float)rows);
    }

    // ================ Cheby layer 3 @ V=1280, F=32 -> 3 (no BN/relu) ======
    {
        const int N = BATCH * 32, N4 = N / 4;
        size_t rows = (size_t)V2 * BATCH;
        k_spmm <<<cdiv((size_t)V2 * N4, T), blk, 0, stream>>>(L2c, L2v, F3, F1, V2, N4);
        k_spmm2<<<cdiv((size_t)V2 * N4, T), blk, 0, stream>>>(L2c, L2v, F1, F3, F0, V2, N4);
        k_build_xs<<<cdiv(rows * 96, T), blk, 0, stream>>>(F3, F1, F0, Abf, rows, 32);
        k_convert_pad_w<<<cdiv((size_t)16 * 96, T), blk, 0, stream>>>(cl3_w, Wbf, 3, 16, 96);
        // out_mode=1 scatters to [B,1280,3] order directly into d_out
        k_gemm_bf16_wmma<<<dim3((int)(rows / 128), 1), blk, 0, stream>>>(
            Abf, Wbf, cl3_b, out, (int)rows, 3, 96, 0, 1, V2, BATCH);
    }
}
